// SequenceMemoryUpdater_91225105367139
// MI455X (gfx1250) — compile-verified, matrix operand order checked by reading, not run on hardware
//
#include <hip/hip_runtime.h>
#include <hip/hip_bf16.h>

// ---------------------------------------------------------------------------
// TGN memory updater: gather -> GRUCell (bf16 WMMA, f32 accum) -> LayerNorm
// -> scatter. Bulk memory copy via async D2D memcpy (bandwidth-bound).
// Weights converted to bf16 once per call into d_ws so the GEMM inner loop is
// pure b128 loads + v_wmma_f32_16x16x32_bf16 (no per-WMMA conversion VALU).
// Target: gfx1250 (CDNA5), wave32.
// ---------------------------------------------------------------------------

typedef __attribute__((ext_vector_type(16))) __bf16 v16bf;
typedef __attribute__((ext_vector_type(8)))  float  v8f;

#define MSG_DIM 512
#define MEM_DIM 256
#define KCAT    (MSG_DIM + MEM_DIM)   // 768 combined A width in LDS
#define GATES   (3 * MEM_DIM)         // 768 gate columns
#define ROWS    16                    // nodes per block
#define NWAVES  8
#define WCOLS   (GATES / NWAVES)      // 96 columns per wave
#define CTILES  (WCOLS / 16)          // 6 column tiles per wave

__device__ __forceinline__ unsigned pack_bf16x2(float a, float b) {
#if __has_builtin(__builtin_amdgcn_cvt_pk_bf16_f32)
    typedef __attribute__((ext_vector_type(2))) __bf16 v2bf;
    v2bf v = __builtin_amdgcn_cvt_pk_bf16_f32(a, b);
    return __builtin_bit_cast(unsigned, v);
#else
    unsigned ua = __float_as_uint(a);
    unsigned ub = __float_as_uint(b);
    ua = (ua + 0x7FFFu + ((ua >> 16) & 1u)) >> 16;   // RNE f32 -> bf16
    ub = (ub + 0x7FFFu + ((ub >> 16) & 1u)) >> 16;
    return ua | (ub << 16);
#endif
}

__device__ __forceinline__ float sigmoidf_(float x) {
    return 1.0f / (1.0f + __expf(-x));
}

struct GateBufs {
    float r  [ROWS * MEM_DIM];   // r gate, later reused to hold new_h
    float z  [ROWS * MEM_DIM];
    float in_[ROWS * MEM_DIM];   // i_n (later reused for LN reduction)
    float hn [ROWS * MEM_DIM];   // h_n (later reused for LN stats)
};

union Smem {
    __bf16  a[ROWS * KCAT];      // staged [msg | h] in bf16 (phase A/GEMM)
    GateBufs b;                  // gate values (phase B/C)
};

// ---------------------------------------------------------------------------
// f32 -> bf16 weight conversion (once per call, into d_ws)
// ---------------------------------------------------------------------------
__global__ void __launch_bounds__(256)
cvt_f32_to_bf16_kernel(const float* __restrict__ src, __bf16* __restrict__ dst,
                       int n_pairs) {
    int stride = gridDim.x * 256;
    for (int i = blockIdx.x * 256 + threadIdx.x; i < n_pairs; i += stride) {
        float2 f = *(const float2*)(src + 2 * (size_t)i);
        *(unsigned*)(dst + 2 * (size_t)i) = pack_bf16x2(f.x, f.y);
    }
}

// ---------------------------------------------------------------------------
// Fragment loaders
// ---------------------------------------------------------------------------
// 16x32 bf16 A fragment from LDS (row-major, stride KCAT). Per ISA layout:
// lanes 0-15 -> M=lane, Kbase=0; lanes 16-31 -> M=lane-16, Kbase=8.
// VGPRs 0..3: K=Kbase+0..7; VGPRs 4..7: K=16+Kbase+0..7 (two b128 loads).
__device__ __forceinline__ v16bf load_a_frag(const __bf16* abase, int kk, int lane) {
    int row = lane & 15;
    int kb  = (lane >> 4) * 8;
    const __bf16* p = abase + row * KCAT + kk + kb;
    union { v16bf bf; int4 q[2]; } f;
    f.q[0] = *(const int4*)(p);
    f.q[1] = *(const int4*)(p + 16);
    return f.bf;
}

// 32x16 B fragment for G = X @ W^T from bf16 weights (row-major, stride ldk).
// Column n of B is row n of W; same lane/K layout as A. Pure b128 loads.
__device__ __forceinline__ v16bf load_b_frag_bf16(const __bf16* W, int ldk,
                                                  int col0, int kk, int lane) {
    int n  = col0 + (lane & 15);
    int kb = (lane >> 4) * 8;
    const __bf16* p = W + (size_t)n * ldk + kk + kb;
    union { v16bf bf; int4 q[2]; } f;
    f.q[0] = *(const int4*)(p);
    f.q[1] = *(const int4*)(p + 16);
    return f.bf;
}

// Fallback: B fragment straight from f32 weights (conversion in-register).
__device__ __forceinline__ v16bf load_b_frag_f32(const float* W, int ldk,
                                                 int col0, int kk, int lane) {
    int n  = col0 + (lane & 15);
    int kb = (lane >> 4) * 8;
    const float* p = W + (size_t)n * ldk + kk + kb;
    float4 f0 = *(const float4*)(p);
    float4 f1 = *(const float4*)(p + 4);
    float4 f2 = *(const float4*)(p + 16);
    float4 f3 = *(const float4*)(p + 20);
    union { v16bf bf; unsigned u[8]; } f;
    f.u[0] = pack_bf16x2(f0.x, f0.y);
    f.u[1] = pack_bf16x2(f0.z, f0.w);
    f.u[2] = pack_bf16x2(f1.x, f1.y);
    f.u[3] = pack_bf16x2(f1.z, f1.w);
    f.u[4] = pack_bf16x2(f2.x, f2.y);
    f.u[5] = pack_bf16x2(f2.z, f2.w);
    f.u[6] = pack_bf16x2(f3.x, f3.y);
    f.u[7] = pack_bf16x2(f3.z, f3.w);
    return f.bf;
}

// ---------------------------------------------------------------------------
// Fused GRU + LayerNorm + scatter
// ---------------------------------------------------------------------------
template <bool WBF16>
__global__ void __launch_bounds__(256)
gru_ln_scatter_kernel(const int*   __restrict__ ids,
                      const float* __restrict__ msgs,
                      const float* __restrict__ ts,
                      const float* __restrict__ memory,
                      const void*  __restrict__ Wih_p,
                      const void*  __restrict__ Whh_p,
                      const float* __restrict__ b_ih,
                      const float* __restrict__ b_hh,
                      const float* __restrict__ ln_gamma,
                      const float* __restrict__ ln_beta,
                      float*       __restrict__ out_mem,
                      float*       __restrict__ out_last,
                      int n_upd)
{
    __shared__ Smem sm;

    const int t    = threadIdx.x;
    const int lane = t & 31;
    const int wave = t >> 5;
    const int row0 = blockIdx.x * ROWS;

    // ---------------- Phase A: stage [msg | h_gathered] as bf16 into LDS ----
    {
        const int pairs = ROWS * KCAT / 2;       // 6144 dword writes
        for (int p = t; p < pairs; p += 256) {
            int e   = p * 2;
            int row = e / KCAT;
            int col = e - row * KCAT;
            int g   = row0 + row;
            if (g >= n_upd) g = n_upd - 1;       // clamp (safe re-read)
            float f0, f1;
            if (col < MSG_DIM) {
                const float* mp = msgs + (size_t)g * MSG_DIM + col;
                f0 = mp[0]; f1 = mp[1];
            } else {
                int id = ids[g];
                const float* hp = memory + (size_t)id * MEM_DIM + (col - MSG_DIM);
                f0 = hp[0]; f1 = hp[1];
            }
            *(unsigned*)(&sm.a[e]) = pack_bf16x2(f0, f1);
        }
    }
    __syncthreads();

    // ---------------- Phase GEMM: gi = X@W_ih^T, gh = h@W_hh^T -------------
    const int col0w = wave * WCOLS;

    v8f acc_gi[CTILES];
    v8f acc_gh[CTILES];
#pragma unroll
    for (int ct = 0; ct < CTILES; ++ct) { acc_gi[ct] = {}; acc_gh[ct] = {}; }

    // gi: K = 512 over messages (LDS cols [0,512))
#pragma unroll 4
    for (int kk = 0; kk < MSG_DIM; kk += 32) {
        v16bf af = load_a_frag(sm.a, kk, lane);
#pragma unroll
        for (int ct = 0; ct < CTILES; ++ct) {
            v16bf bf = WBF16
                ? load_b_frag_bf16((const __bf16*)Wih_p, MSG_DIM, col0w + ct * 16, kk, lane)
                : load_b_frag_f32 ((const float*) Wih_p, MSG_DIM, col0w + ct * 16, kk, lane);
            acc_gi[ct] = __builtin_amdgcn_wmma_f32_16x16x32_bf16(
                false, af, false, bf, (short)0, acc_gi[ct], false, false);
        }
    }
    // gh: K = 256 over gathered h (LDS cols [512,768))
#pragma unroll 4
    for (int kk = 0; kk < MEM_DIM; kk += 32) {
        v16bf af = load_a_frag(sm.a, MSG_DIM + kk, lane);
#pragma unroll
        for (int ct = 0; ct < CTILES; ++ct) {
            v16bf bf = WBF16
                ? load_b_frag_bf16((const __bf16*)Whh_p, MEM_DIM, col0w + ct * 16, kk, lane)
                : load_b_frag_f32 ((const float*) Whh_p, MEM_DIM, col0w + ct * 16, kk, lane);
            acc_gh[ct] = __builtin_amdgcn_wmma_f32_16x16x32_bf16(
                false, af, false, bf, (short)0, acc_gh[ct], false, false);
        }
    }
    __syncthreads();   // 'a' staging buffer dead; gate buffers go live

    // ---------------- Phase B: biases + gate nonlinearities into LDS -------
    // C/D layout: VGPR i, lanes 0-15 -> M=i, N=lane; lanes 16-31 -> M=i+8.
    {
        int lrow8 = (lane >> 4) * 8;
        int lcol  = lane & 15;
#pragma unroll
        for (int ct = 0; ct < CTILES; ++ct) {
            int col0 = col0w + ct * 16;
            int c    = col0 + lcol;
            int gate = col0 >> 8;                 // 0=r, 1=z, 2=n (tile-uniform)
            float bi = b_ih[c];
            float bh = b_hh[c];
#pragma unroll
            for (int i = 0; i < 8; ++i) {
                int row  = i + lrow8;
                float vg = acc_gi[ct][i] + bi;
                float vh = acc_gh[ct][i] + bh;
                if (gate == 0) {
                    sm.b.r[row * MEM_DIM + c] = sigmoidf_(vg + vh);
                } else if (gate == 1) {
                    sm.b.z[row * MEM_DIM + (c - MEM_DIM)] = sigmoidf_(vg + vh);
                } else {
                    sm.b.in_[row * MEM_DIM + (c - 2 * MEM_DIM)] = vg;
                    sm.b.hn [row * MEM_DIM + (c - 2 * MEM_DIM)] = vh;
                }
            }
        }
    }
    __syncthreads();

    // ---------------- Phase C: GRU combine + LayerNorm + scatter -----------
    const int row = t >> 4;               // 16 threads per node row
    const int cs  = (t & 15) * 16;        // each thread: 16 hidden columns
    const int g   = row0 + row;
    const bool valid = (g < n_upd);
    const int id  = valid ? ids[g] : 0;
    const float* hold = memory + (size_t)id * MEM_DIM;

    float sum = 0.0f, sumsq = 0.0f;
#pragma unroll
    for (int j = 0; j < 16; ++j) {
        int c     = cs + j;
        float rv  = sm.b.r  [row * MEM_DIM + c];
        float zv  = sm.b.z  [row * MEM_DIM + c];
        float inv = sm.b.in_[row * MEM_DIM + c];
        float hnv = sm.b.hn [row * MEM_DIM + c];
        float nv  = tanhf(inv + rv * hnv);
        float h   = hold[c];
        float hv  = (1.0f - zv) * nv + zv * h;
        sm.b.r[row * MEM_DIM + c] = hv;   // reuse r buf for new_h (exclusive slot)
        sum   += hv;
        sumsq += hv * hv;
    }
    __syncthreads();

    float* red = sm.b.in_;                // reuse (dead) for LN reduction
    red[t]       = sum;
    red[256 + t] = sumsq;
    __syncthreads();

    float* stats = sm.b.hn;               // reuse (dead) for mu / rstd
    if (t < ROWS) {
        float s = 0.0f, ss = 0.0f;
#pragma unroll
        for (int j = 0; j < 16; ++j) {
            s  += red[t * 16 + j];
            ss += red[256 + t * 16 + j];
        }
        float mu  = s * (1.0f / (float)MEM_DIM);
        float var = ss * (1.0f / (float)MEM_DIM) - mu * mu;
        stats[t]        = mu;
        stats[ROWS + t] = rsqrtf(var + 1e-5f);
    }
    __syncthreads();

    if (valid) {
        float mu   = stats[row];
        float rstd = stats[ROWS + row];
        float* op  = out_mem + (size_t)id * MEM_DIM;
#pragma unroll
        for (int j = 0; j < 16; ++j) {
            int c    = cs + j;
            float hv = sm.b.r[row * MEM_DIM + c];
            op[c] = (hv - mu) * rstd * ln_gamma[c] + ln_beta[c];
        }
        if ((t & 15) == 0) out_last[id] = ts[g];
    }
}

extern "C" void kernel_launch(void* const* d_in, const int* in_sizes, int n_in,
                              void* d_out, int out_size, void* d_ws, size_t ws_size,
                              hipStream_t stream) {
    const int*   ids      = (const int*)  d_in[0];
    const float* msgs     = (const float*)d_in[1];
    const float* ts       = (const float*)d_in[2];
    const float* memory   = (const float*)d_in[3];
    const float* last_upd = (const float*)d_in[4];
    const float* W_ih     = (const float*)d_in[5];
    const float* W_hh     = (const float*)d_in[6];
    const float* b_ih     = (const float*)d_in[7];
    const float* b_hh     = (const float*)d_in[8];
    const float* ln_gamma = (const float*)d_in[9];
    const float* ln_beta  = (const float*)d_in[10];

    const int n_upd   = in_sizes[0];
    const int n_nodes = in_sizes[4];

    float* out_mem  = (float*)d_out;
    float* out_last = out_mem + (size_t)n_nodes * MEM_DIM;

    // Bulk copy (bandwidth-bound, ~1 GB total): memory -> out, last_update -> out.
    hipMemcpyAsync(out_mem, memory, (size_t)n_nodes * MEM_DIM * sizeof(float),
                   hipMemcpyDeviceToDevice, stream);
    hipMemcpyAsync(out_last, last_upd, (size_t)n_nodes * sizeof(float),
                   hipMemcpyDeviceToDevice, stream);

    const size_t wih_elems = (size_t)GATES * MSG_DIM;   // 393216
    const size_t whh_elems = (size_t)GATES * MEM_DIM;   // 196608
    const size_t wbytes    = (wih_elems + whh_elems) * sizeof(__bf16);   // ~1.13 MB

    const int nblocks = (n_upd + ROWS - 1) / ROWS;

    if (ws_size >= wbytes) {
        __bf16* wih_bf = (__bf16*)d_ws;
        __bf16* whh_bf = wih_bf + wih_elems;
        cvt_f32_to_bf16_kernel<<<(int)(wih_elems / 2 / 256), 256, 0, stream>>>(
            W_ih, wih_bf, (int)(wih_elems / 2));
        cvt_f32_to_bf16_kernel<<<(int)(whh_elems / 2 / 256), 256, 0, stream>>>(
            W_hh, whh_bf, (int)(whh_elems / 2));
        gru_ln_scatter_kernel<true><<<nblocks, 256, 0, stream>>>(
            ids, msgs, ts, memory, wih_bf, whh_bf, b_ih, b_hh,
            ln_gamma, ln_beta, out_mem, out_last, n_upd);
    } else {
        gru_ln_scatter_kernel<false><<<nblocks, 256, 0, stream>>>(
            ids, msgs, ts, memory, W_ih, W_hh, b_ih, b_hh,
            ln_gamma, ln_beta, out_mem, out_last, n_upd);
    }
}
